// GraphAttentionLayer_21294447853884
// MI455X (gfx1250) — matrix operand order repaired and checked by reading, not compiled, not run on hardware
//
#include <hip/hip_runtime.h>

#define N_NODES 8192
#define F 64
#define ALPHA 0.2f
#define NEG_INF (-__builtin_inff())

typedef __attribute__((ext_vector_type(16))) _Float16 v16h;
typedef __attribute__((ext_vector_type(8)))  float    v8f;

// ---------------------------------------------------------------------------
// Kernel 1: Wh = h @ W^T   (8192x64, fp32).  W transposed into LDS so lanes
// read consecutive banks; h row address is wave-uniform (scalarized).
// ---------------------------------------------------------------------------
__global__ __launch_bounds__(256) void gat_wh(const float* __restrict__ h,
                                              const float* __restrict__ W,
                                              float* __restrict__ Wh) {
    __shared__ float WT[F * F];  // WT[k*F + f] = W[f*F + k]
    int t = threadIdx.x;
    for (int idx = t; idx < F * F; idx += 256) {
        int f = idx >> 6, k = idx & 63;
        WT[k * F + f] = W[f * F + k];
    }
    __syncthreads();
    int row = blockIdx.x * 4 + (t >> 6);
    int f   = t & 63;
    const float* hr = h + (long)row * F;
    float acc = 0.f;
#pragma unroll
    for (int k = 0; k < F; ++k) acc = fmaf(hr[k], WT[k * F + f], acc);
    Wh[(long)row * F + f] = acc;
}

// ---------------------------------------------------------------------------
// Kernel 2: src[i] = Wh[i,:]·a1 ; dst[i] = Wh[i,:]·a2
// ---------------------------------------------------------------------------
__global__ __launch_bounds__(256) void gat_srcdst(const float* __restrict__ Wh,
                                                  const float* __restrict__ a,
                                                  float* __restrict__ srcv,
                                                  float* __restrict__ dstv) {
    int i = blockIdx.x * 256 + threadIdx.x;
    const float* w = Wh + (long)i * F;
    float s = 0.f, d = 0.f;
#pragma unroll
    for (int k = 0; k < F; ++k) {
        s = fmaf(w[k], a[k], s);
        d = fmaf(w[k], a[F + k], d);
    }
    srcv[i] = s;
    dstv[i] = d;
}

// ---------------------------------------------------------------------------
// Kernel 3: maxdst = max_j dst[j]  (single block reduction).
// Gives a guaranteed upper bound m_i = lrelu(src_i + maxdst) >= every score
// of row i, so exp(e - m_i) <= 1: softmax needs no online max tracking.
// ---------------------------------------------------------------------------
__global__ __launch_bounds__(256) void gat_maxdst(const float* __restrict__ dstv,
                                                  float* __restrict__ maxout) {
    __shared__ float red[256];
    float m = NEG_INF;
    for (int i = threadIdx.x; i < N_NODES; i += 256) m = fmaxf(m, dstv[i]);
    red[threadIdx.x] = m;
    __syncthreads();
    for (int s = 128; s > 0; s >>= 1) {
        if (threadIdx.x < s) red[threadIdx.x] = fmaxf(red[threadIdx.x], red[threadIdx.x + s]);
        __syncthreads();
    }
    if (threadIdx.x == 0) maxout[0] = red[0];
}

// ---------------------------------------------------------------------------
// Kernel 4: pack Wh into f16 WMMA B-matrix layout (32x16 tiles, K-major per
// lane): Bp[((jblock*4+ftile)*32+lane)*16 + kidx] = Wh[jblock*32 + khalf +
// (kidx>=8?16:0) + kidx%8, ftile*16 + lane%16], khalf = (lane>=16)?8:0.  One contiguous 32B
// run per lane fragment in the hot loop (L2-resident, 1 MB).
// ---------------------------------------------------------------------------
__global__ __launch_bounds__(256) void gat_bpack(const float* __restrict__ Wh,
                                                 _Float16* __restrict__ Bp) {
    int e = blockIdx.x * 256 + threadIdx.x;  // e in [0, N*F)
    int kidx = e & 15;
    int lane = (e >> 4) & 31;
    int ft   = (e >> 9) & 3;
    int jb   = e >> 11;
    int j = jb * 32 + ((kidx & 8) ? 16 : 0) + ((lane & 16) ? 8 : 0) + (kidx & 7);
    int f = ft * 16 + (lane & 15);
    Bp[e] = (_Float16)Wh[(long)j * F + f];
}

// ---------------------------------------------------------------------------
// Kernel 5: fused masked-softmax attention + P@Wh via WMMA.
// Grid: 512 blocks (one 16-row tile each), 256 threads = 8 waves.
// Each wave owns j in [wave*1024, wave*1024+1024); fixed exp bound means the
// loop is pure stream->exp->wmma with no cross-lane traffic.
// Per iteration: B fragments loaded first into DISTINCT registers (single
// base + immediate offsets), latency hidden by the exp/cvt block; then the
// four WMMAs issue back-to-back with no WAR register reuse between them.
// ---------------------------------------------------------------------------
__global__ __launch_bounds__(256) void gat_attn(const int* __restrict__ adj,
                                                const float* __restrict__ srcv,
                                                const float* __restrict__ dstv,
                                                const float* __restrict__ maxdst,
                                                const _Float16* __restrict__ Bp,
                                                float* __restrict__ out) {
    __shared__ float accS[8][16][64];
    __shared__ float sS[8][16];

    const int lane  = threadIdx.x & 31;
    const int wave  = threadIdx.x >> 5;
    const int i0    = blockIdx.x * 16;
    const int myrow = lane & 15;
    const int koff  = (lane < 16) ? 0 : 8;  // 16-bit A layout K-half
    const int cbase = (lane < 16) ? 0 : 8;  // C rows this lane holds
    const int j0    = wave * (N_NODES / 8);
    const int ITER  = (N_NODES / 8) / 32;

    const float si   = srcv[i0 + myrow];
    const float tmax = si + maxdst[0];
    const float mhat = fmaxf(tmax, ALPHA * tmax);  // >= every score in the row
    const long  arow = (long)(i0 + myrow) * N_NODES;

    // per-lane B stream base: fragment ft of jblock jb lives at
    //   Bp + (jb*128 + lane)*16 + ft*512   (halfs); advance 2048 halfs/iter
    const _Float16* bptr = Bp + ((long)(j0 >> 5) * 128 + lane) * 16;

    float ps0 = 0.f, ps1 = 0.f, ps2 = 0.f, ps3 = 0.f;  // independent partials
    v8f acc[4] = {};

    for (int it = 0; it < ITER; ++it) {
        const int jbase = j0 + it * 32;
        const int run1  = jbase + koff;

        // ---- issue all loads first: 4 distinct B fragments (L2 hits,
        //      constant immediate offsets off one base), then adj + dst ----
        v16h B0 = *(const v16h*)(bptr);
        v16h B1 = *(const v16h*)(bptr + 512);
        v16h B2 = *(const v16h*)(bptr + 1024);
        v16h B3 = *(const v16h*)(bptr + 1536);
        int4 a0 = *(const int4*)(adj + arow + run1);
        int4 a1 = *(const int4*)(adj + arow + run1 + 4);
        int4 a2 = *(const int4*)(adj + arow + run1 + 16);
        int4 a3 = *(const int4*)(adj + arow + run1 + 20);
        float4 d0 = *(const float4*)(dstv + run1);
        float4 d1 = *(const float4*)(dstv + run1 + 4);
        float4 d2 = *(const float4*)(dstv + run1 + 16);
        float4 d3 = *(const float4*)(dstv + run1 + 20);
        if (it + 1 < ITER)
            __builtin_prefetch(adj + arow + run1 + 32, 0, 3);  // global_prefetch_b8
        bptr += 2048;

        int   av[16] = {a0.x, a0.y, a0.z, a0.w, a1.x, a1.y, a1.z, a1.w,
                        a2.x, a2.y, a2.z, a2.w, a3.x, a3.y, a3.z, a3.w};
        float dv[16] = {d0.x, d0.y, d0.z, d0.w, d1.x, d1.y, d1.z, d1.w,
                        d2.x, d2.y, d2.z, d2.w, d3.x, d3.y, d3.z, d3.w};

        // p_k = exp(leaky_relu(src+dst) - mhat), masked -> exp(-inf) = 0
        // (this VALU/TRANS block hides the B-fragment load latency)
        float p[16];
#pragma unroll
        for (int k = 0; k < 16; ++k) {
            float t   = si + dv[k];
            float lr  = fmaxf(t, ALPHA * t);                  // leaky_relu
            float arg = (av[k] > 0) ? (lr - mhat) : NEG_INF;  // <= 0 always
            p[k] = __expf(arg);
        }
        // four independent sum chains (no serial VALU dependency)
        ps0 += p[0] + p[4] + p[8]  + p[12];
        ps1 += p[1] + p[5] + p[9]  + p[13];
        ps2 += p[2] + p[6] + p[10] + p[14];
        ps3 += p[3] + p[7] + p[11] + p[15];

        // f16 A tile, already in WMMA 16-bit A-layout order for this lane
        v16h A;
#pragma unroll
        for (int k = 0; k < 16; ++k) A[k] = (_Float16)p[k];

        // 4 back-to-back WMMAs: P(16x32,f16) x Wh(32x16,f16) += acc
        // (distinct B registers -> no WAR hazards between them)
        acc[0] = __builtin_amdgcn_wmma_f32_16x16x32_f16(false, A, false, B0,
                                                        (short)0, acc[0], false, false);
        acc[1] = __builtin_amdgcn_wmma_f32_16x16x32_f16(false, A, false, B1,
                                                        (short)0, acc[1], false, false);
        acc[2] = __builtin_amdgcn_wmma_f32_16x16x32_f16(false, A, false, B2,
                                                        (short)0, acc[2], false, false);
        acc[3] = __builtin_amdgcn_wmma_f32_16x16x32_f16(false, A, false, B3,
                                                        (short)0, acc[3], false, false);
    }

    float ssum = (ps0 + ps1) + (ps2 + ps3);
    ssum += __shfl_xor(ssum, 16, 32);  // combine the two K-halves of each row

    // dump wave partials: C layout -> row = cbase+r, col = ft*16 + lane%16
    const int col0 = lane & 15;
#pragma unroll
    for (int ft = 0; ft < 4; ++ft)
#pragma unroll
        for (int r = 0; r < 8; ++r)
            accS[wave][cbase + r][ft * 16 + col0] = acc[ft][r];
    if (lane < 16) sS[wave][lane] = ssum;
    __syncthreads();

    // merge 8 wave partials (same exp bound everywhere -> plain sums),
    // normalize, ELU, store fp32
    for (int o = threadIdx.x; o < 16 * 64; o += 256) {
        int row = o >> 6, col = o & 63;
        float S = 0.f, val = 0.f;
#pragma unroll
        for (int w = 0; w < 8; ++w) {
            S   += sS[w][row];
            val += accS[w][row][col];
        }
        float hp = val / S;
        float y  = hp > 0.f ? hp : (__expf(hp) - 1.f);  // elu (alpha=1)
        out[(long)(i0 + row) * F + col] = y;
    }
}

// ---------------------------------------------------------------------------
// Host wiring
// ---------------------------------------------------------------------------
extern "C" void kernel_launch(void* const* d_in, const int* in_sizes, int n_in,
                              void* d_out, int out_size, void* d_ws, size_t ws_size,
                              hipStream_t stream) {
    const float* h   = (const float*)d_in[0];
    const int*   adj = (const int*)d_in[1];
    const float* W   = (const float*)d_in[2];
    const float* a   = (const float*)d_in[3];
    float* out = (float*)d_out;

    // workspace: Wh fp32 | src | dst | maxdst | Bpacked f16
    float* Wh     = (float*)d_ws;              // N*F floats (2 MB)
    float* srcv   = Wh + (long)N_NODES * F;    // N floats
    float* dstv   = srcv + N_NODES;            // N floats
    float* maxdst = dstv + N_NODES;            // 1 float (padded to 16)
    _Float16* Bp  = (_Float16*)(maxdst + 16);  // N*F halfs (1 MB)

    gat_wh    <<<N_NODES / 4,         256, 0, stream>>>(h, W, Wh);
    gat_srcdst<<<N_NODES / 256,       256, 0, stream>>>(Wh, a, srcv, dstv);
    gat_maxdst<<<1,                   256, 0, stream>>>(dstv, maxdst);
    gat_bpack <<<(N_NODES * F) / 256, 256, 0, stream>>>(Wh, Bp);
    gat_attn  <<<N_NODES / 16,        256, 0, stream>>>(adj, srcv, dstv, maxdst, Bp, out);
}
